// LSTMLayer_56049323213027
// MI455X (gfx1250) — matrix operand order repaired and checked by reading, not compiled, not run on hardware
//
#include <hip/hip_runtime.h>
#include <hip/hip_bf16.h>

// ---------------- problem constants ----------------
#define BB   64          // batch
#define TT   512         // time steps
#define DD   512         // input dim
#define NN   1024        // hidden dim
#define GG   4096        // 4*N gate dim
#define BT   32768       // B*T rows of x
// tiles
#define KT_X 16          // DD/32
#define KT_H 32          // NN/32
#define NTIL 256         // GG/16

typedef __attribute__((ext_vector_type(16))) __bf16 v16bf;
typedef __attribute__((ext_vector_type(8)))  float  v8f;

static __device__ __forceinline__ unsigned short f2bf(float f) {
    unsigned int u = __float_as_uint(f);
    unsigned int r = (u + 0x7FFFu + ((u >> 16) & 1u)) >> 16;
    return (unsigned short)r;
}

static __device__ __forceinline__ v16bf make_frag(uint4 lo, uint4 hi) {
    union { v16bf v; uint4 q[2]; } u;
    u.q[0] = lo; u.q[1] = hi;
    return u.v;
}

// A-fragment from row-major bf16 (row stride = ld elems): lane holds two
// 16-byte chunks: K..K+7 and K+16..K+23 (kSel pre-added into k0)
static __device__ __forceinline__ v16bf loadA(const unsigned short* rowbase, int k0) {
    const uint4* p = reinterpret_cast<const uint4*>(rowbase + k0);
    return make_frag(p[0], p[2]);
}
// B-fragment from packed tile: 32 contiguous bytes per lane
static __device__ __forceinline__ v16bf loadB(const unsigned short* laneptr) {
    const uint4* p = reinterpret_cast<const uint4*>(laneptr);
    return make_frag(p[0], p[1]);
}

static __device__ __forceinline__ float sigm(float x) {
    return 1.0f / (1.0f + __expf(-x));
}

#define WMMA_BF16(A, B, C) \
    __builtin_amdgcn_wmma_f32_16x16x32_bf16(false, (A), false, (B), (short)0, (C), false, false)

// ---------------- pack W into WMMA B-fragment layout (bf16) ----------------
// element j (0..511) of a 32x16 tile: lane = j/16, jj = j%16
//   -> n = nt*16 + (lane&15), k = kt*32 + (jj>=8 ? 16 : 0) + (lane>=16 ? 8 : 0) + (jj&7)
__global__ void k_pack_w(const float* __restrict__ W,
                         unsigned short* __restrict__ wxp,
                         unsigned short* __restrict__ whp) {
    const int WX = KT_X * NTIL * 512;
    const int WH = KT_H * NTIL * 512;
    int stride = gridDim.x * blockDim.x;
    for (int p = blockIdx.x * blockDim.x + threadIdx.x; p < WX + WH; p += stride) {
        bool isWh = (p >= WX);
        int q    = isWh ? (p - WX) : p;
        int tile = q >> 9;
        int j    = q & 511;
        int lane = j >> 4;
        int jj   = j & 15;
        int nt   = tile & 255;
        int kt   = tile >> 8;
        int k    = kt * 32 + ((jj >> 3) << 4) + ((lane >> 4) << 3) + (jj & 7);
        int n    = nt * 16 + (lane & 15);
        int krow = isWh ? (DD + k) : k;
        unsigned short b = f2bf(W[(size_t)krow * GG + n]);
        if (isWh) whp[q] = b; else wxp[q] = b;
    }
}

// ---------------- x (f32) -> bf16 row-major ----------------
__global__ void k_conv_x(const float* __restrict__ x, unsigned short* __restrict__ xbf) {
    int stride = gridDim.x * blockDim.x;
    const int NTOT = BT * DD;
    for (int i = blockIdx.x * blockDim.x + threadIdx.x; i < NTOT; i += stride)
        xbf[i] = f2bf(x[i]);
}

// ---------------- zero c and h0 ----------------
__global__ void k_init(float* __restrict__ c, unsigned short* __restrict__ h0) {
    int stride = gridDim.x * blockDim.x;
    const int NTOT = BB * NN;
    for (int i = blockIdx.x * blockDim.x + threadIdx.x; i < NTOT; i += stride) {
        c[i] = 0.0f;
        h0[i] = 0;
    }
}

// ---------------- GEMM1: x_proj = x_bf16 @ Wx + bias  (32768x512x4096) ----
// one wave: 2 m-tiles x 4 n-tiles, software-pipelined K loop (ping-pong regs),
// LDS-staged coalesced b128 writeback.
__global__ void __launch_bounds__(256)
k_gemm1(const unsigned short* __restrict__ xbf,
        const unsigned short* __restrict__ wxp,
        const float* __restrict__ bias,
        float* __restrict__ xproj) {
    __shared__ float sc[8 * 16 * 64];    // 8 waves * (16x64 f32 block) = 32 KB

    int tid  = threadIdx.x;
    int lane = tid & 31;
    int wv   = tid >> 5;
    int w    = blockIdx.x * 8 + wv;
    int mp   = w >> 6;            // m-pair 0..1023 (2 m-tiles each)
    int ng   = w & 63;            // group of 4 n-tiles
    int kSel = (lane >> 4) << 3;  // 0 or 8

    const unsigned short* arow0 = xbf + (size_t)(mp * 32 + (lane & 15)) * DD;
    const unsigned short* arow1 = arow0 + (size_t)16 * DD;
    const unsigned short* bbase = wxp + (size_t)(ng * 4) * 512 + lane * 16;
    const size_t bstride = (size_t)NTIL * 512;   // per-kt stride in elems

    v8f z = {0.f,0.f,0.f,0.f,0.f,0.f,0.f,0.f};
    v8f acc[2][4] = {{z,z,z,z},{z,z,z,z}};

    v16bf a0C, a1C, b0C, b1C, b2C, b3C;
    v16bf a0N, a1N, b0N, b1N, b2N, b3N;

    auto LD = [&](int kt, v16bf& A0, v16bf& A1,
                  v16bf& B0, v16bf& B1, v16bf& B2, v16bf& B3) {
        int k0 = kt * 32 + kSel;
        A0 = loadA(arow0, k0);
        A1 = loadA(arow1, k0);
        const unsigned short* bp = bbase + (size_t)kt * bstride;
        B0 = loadB(bp);
        B1 = loadB(bp + 512);
        B2 = loadB(bp + 1024);
        B3 = loadB(bp + 1536);
    };

    LD(0, a0C, a1C, b0C, b1C, b2C, b3C);
    #pragma unroll
    for (int kt = 0; kt < KT_X; kt += 2) {
        LD(kt + 1, a0N, a1N, b0N, b1N, b2N, b3N);
        acc[0][0] = WMMA_BF16(a0C, b0C, acc[0][0]);
        acc[0][1] = WMMA_BF16(a0C, b1C, acc[0][1]);
        acc[0][2] = WMMA_BF16(a0C, b2C, acc[0][2]);
        acc[0][3] = WMMA_BF16(a0C, b3C, acc[0][3]);
        acc[1][0] = WMMA_BF16(a1C, b0C, acc[1][0]);
        acc[1][1] = WMMA_BF16(a1C, b1C, acc[1][1]);
        acc[1][2] = WMMA_BF16(a1C, b2C, acc[1][2]);
        acc[1][3] = WMMA_BF16(a1C, b3C, acc[1][3]);
        if (kt + 2 < KT_X) LD(kt + 2, a0C, a1C, b0C, b1C, b2C, b3C);
        acc[0][0] = WMMA_BF16(a0N, b0N, acc[0][0]);
        acc[0][1] = WMMA_BF16(a0N, b1N, acc[0][1]);
        acc[0][2] = WMMA_BF16(a0N, b2N, acc[0][2]);
        acc[0][3] = WMMA_BF16(a0N, b3N, acc[0][3]);
        acc[1][0] = WMMA_BF16(a1N, b0N, acc[1][0]);
        acc[1][1] = WMMA_BF16(a1N, b1N, acc[1][1]);
        acc[1][2] = WMMA_BF16(a1N, b2N, acc[1][2]);
        acc[1][3] = WMMA_BF16(a1N, b3N, acc[1][3]);
    }

    // C layout: VGPR r, lanes0-15: (M=r, N=lane); lanes16-31: (M=8+r, N=lane-16)
    float* my = sc + wv * (16 * 64);
    int rowW = (lane >> 4) << 3;
    int colL = lane & 15;
    #pragma unroll
    for (int m = 0; m < 2; ++m) {
        #pragma unroll
        for (int i = 0; i < 4; ++i) {
            float bv = bias[(ng * 4 + i) * 16 + colL];
            #pragma unroll
            for (int r = 0; r < 8; ++r)
                my[(rowW + r) * 64 + i * 16 + colL] = acc[m][i][r] + bv;
        }
        // coalesced writeback: 16 rows x 64 f32 -> b128 stores
        // (DS ops are in-order within a wave; no barrier needed for
        //  wave-private LDS region)
        size_t rowBase = (size_t)(mp * 32 + m * 16);
        float* gbase = xproj + rowBase * GG + (size_t)ng * 64;
        const uint4* mq = reinterpret_cast<const uint4*>(my);
        #pragma unroll
        for (int i2 = 0; i2 < 8; ++i2) {
            int pos = i2 * 32 + lane;       // 256 uint4 slots = 16 rows x 16
            int row = pos >> 4;
            int c4  = pos & 15;
            *reinterpret_cast<uint4*>(gbase + (size_t)row * GG + c4 * 4) = mq[pos];
        }
    }
}

// ---------------- LSTM step: gates = xproj[:,t] + h @ Wh, then pointwise ---
// block = 512 thr = 16 waves: wave wv -> gate g = wv/4, K-slice ks = wv%4.
// grid = 4 (m-tiles) * 64 (n-tiles over N=1024) = 256 blocks.
__global__ void __launch_bounds__(512)
k_step(const unsigned short* __restrict__ whp,
       const float* __restrict__ xproj,
       const unsigned short* __restrict__ hin,
       unsigned short* __restrict__ hout,
       float* __restrict__ cbuf,
       float* __restrict__ out,
       int t) {
    __shared__ float lds[16 * 256];   // 16 partial 16x16 tiles = 16 KB

    int tid  = threadIdx.x;
    int lane = tid & 31;
    int wv   = tid >> 5;      // 0..15
    int ks   = wv & 3;        // K slice
    int g    = wv >> 2;       // gate 0..3 (i,j,f,o)
    int mt   = blockIdx.x >> 6;
    int nt   = blockIdx.x & 63;

    int kSel  = (lane >> 4) << 3;
    int ntile = g * 64 + nt;  // column tile in 0..255 of the 4096-wide gates

    const unsigned short* arow  = hin + (size_t)(mt * 16 + (lane & 15)) * NN;
    const unsigned short* btile = whp + (size_t)ntile * 512 + lane * 16;
    const size_t bstride = (size_t)NTIL * 512;

    v8f acc = {0.f,0.f,0.f,0.f,0.f,0.f,0.f,0.f};
    int kt0 = ks * 8;

    v16bf aC, bC, aN, bN;
    aC = loadA(arow, kt0 * 32 + kSel);
    bC = loadB(btile + (size_t)kt0 * bstride);
    #pragma unroll
    for (int u = 0; u < 8; u += 2) {
        int kt = kt0 + u;
        aN = loadA(arow, (kt + 1) * 32 + kSel);
        bN = loadB(btile + (size_t)(kt + 1) * bstride);
        acc = WMMA_BF16(aC, bC, acc);
        if (u + 2 < 8) {
            aC = loadA(arow, (kt + 2) * 32 + kSel);
            bC = loadB(btile + (size_t)(kt + 2) * bstride);
        }
        acc = WMMA_BF16(aN, bN, acc);
    }

    int row = (lane >> 4) << 3;
    int col = lane & 15;
    #pragma unroll
    for (int r = 0; r < 8; ++r)
        lds[wv * 256 + (row + r) * 16 + col] = acc[r];
    __syncthreads();

    if (tid < 256) {
        int rr = tid >> 4, cc = tid & 15;
        int idx = rr * 16 + cc;
        float gv[4];
        #pragma unroll
        for (int gg = 0; gg < 4; ++gg)
            gv[gg] = lds[(gg * 4 + 0) * 256 + idx] + lds[(gg * 4 + 1) * 256 + idx]
                   + lds[(gg * 4 + 2) * 256 + idx] + lds[(gg * 4 + 3) * 256 + idx];

        int m = mt * 16 + rr;
        int n = nt * 16 + cc;
        size_t xb = ((size_t)m * TT + t) * GG + n;
        float iv = gv[0] + xproj[xb];
        float jv = gv[1] + xproj[xb + 1024];
        float fv = gv[2] + xproj[xb + 2048];
        float ov = gv[3] + xproj[xb + 3072];

        float co = cbuf[m * NN + n];
        float cn = co * sigm(fv + 1.0f) + sigm(iv) * tanhf(jv);
        float hn = tanhf(cn) * sigm(ov);
        cbuf[m * NN + n] = cn;
        out[((size_t)m * TT + t) * NN + n] = hn;
        hout[m * NN + n] = f2bf(hn);
    }
}

// ---------------- host ----------------
extern "C" void kernel_launch(void* const* d_in, const int* in_sizes, int n_in,
                              void* d_out, int out_size, void* d_ws, size_t ws_size,
                              hipStream_t stream) {
    const float* x    = (const float*)d_in[0];
    const float* W    = (const float*)d_in[1];
    const float* bias = (const float*)d_in[2];
    float* out = (float*)d_out;

    char* ws = (char*)d_ws;
    size_t off = 0;
    auto carve = [&](size_t bytes) -> void* {
        off = (off + 255) & ~(size_t)255;
        void* p = ws + off;
        off += bytes;
        return p;
    };
    unsigned short* wxp   = (unsigned short*)carve((size_t)KT_X * NTIL * 512 * 2); //   4 MB
    unsigned short* whp   = (unsigned short*)carve((size_t)KT_H * NTIL * 512 * 2); //   8 MB
    unsigned short* xbf   = (unsigned short*)carve((size_t)BT * DD * 2);           //  32 MB
    float*          xproj = (float*)carve((size_t)BT * GG * 4);                    // 512 MB
    float*          cbuf  = (float*)carve((size_t)BB * NN * 4);
    unsigned short* h0    = (unsigned short*)carve((size_t)BB * NN * 2);
    unsigned short* h1    = (unsigned short*)carve((size_t)BB * NN * 2);
    (void)ws_size; (void)in_sizes; (void)n_in; (void)out_size;

    k_pack_w<<<4096, 256, 0, stream>>>(W, wxp, whp);
    k_conv_x<<<8192, 256, 0, stream>>>(x, xbf);
    k_init<<<256, 256, 0, stream>>>(cbuf, h0);

    // 1024 m-pairs * 64 n-groups = 65536 waves, 8 waves/block
    k_gemm1<<<8192, 256, 0, stream>>>(xbf, wxp, bias, xproj);

    for (int t = 0; t < TT; ++t) {
        const unsigned short* hin = (t & 1) ? h1 : h0;
        unsigned short*      hout = (t & 1) ? h0 : h1;
        k_step<<<256, 512, 0, stream>>>(whp, xproj, hin, hout, cbuf, out, t);
    }
}
